// GNNModel_82171314307241
// MI455X (gfx1250) — compile-verified, matrix operand order checked by reading, not run on hardware
//
#include <hip/hip_runtime.h>
#include <stdint.h>

typedef __attribute__((ext_vector_type(16))) __bf16 v16bf;
typedef __attribute__((ext_vector_type(8)))  float  v8f;

// ---------------- degree / norm precompute ----------------
__global__ __launch_bounds__(256) void k_deg_init(float* __restrict__ deg, int n) {
  int i = blockIdx.x * blockDim.x + threadIdx.x;
  if (i < n) deg[i] = 1.0f;               // self-loop
}

__global__ __launch_bounds__(256) void k_deg_count(const int* __restrict__ dst,
                                                   float* __restrict__ deg, int e) {
  int i = blockIdx.x * blockDim.x + threadIdx.x;
  if (i < e) atomicAdd(&deg[dst[i]], 1.0f);
}

__global__ __launch_bounds__(256) void k_deg_fin(const float* __restrict__ deg,
                                                 float* __restrict__ dinv,
                                                 float* __restrict__ nself, int n) {
  int i = blockIdx.x * blockDim.x + threadIdx.x;
  if (i < n) {
    float d  = deg[i];
    dinv[i]  = rsqrtf(d);
    nself[i] = 1.0f / d;
  }
}

// ---------------- weight pre-pack: f32 [K,F] -> bf16 WMMA B-fragment order ----
// Fragment layout: [ks = K/32][t = F/16][lane 0..31][16 bf16], where for lane l:
//   n = t*16 + (l&15), kb = ks*32 + ((l>>4)<<3),
//   elem i<8  -> W[kb+i][n],  elem i>=8 -> W[kb+16+(i-8)][n]
__global__ __launch_bounds__(256) void k_pack_w(const float* __restrict__ W,
                                                __bf16* __restrict__ Wp,
                                                int K, int F, int total) {
  int gid = blockIdx.x * blockDim.x + threadIdx.x;
  if (gid >= total) return;
  int lane = gid & 31;
  int rest = gid >> 5;
  int nt   = F >> 4;
  int t    = rest % nt;
  int ks   = rest / nt;
  int n    = t * 16 + (lane & 15);
  int kb   = ks * 32 + ((lane >> 4) << 3);
  __bf16* out = Wp + (size_t)gid * 16;
#pragma unroll
  for (int i = 0; i < 8; i++) {
    out[i]     = (__bf16)W[(size_t)(kb + i) * F + n];
    out[8 + i] = (__bf16)W[(size_t)(kb + 16 + i) * F + n];
  }
}

// ---------------- bf16 WMMA GEMM with fused self-loop epilogue ----------------
// Y[N,F] = (RELU? relu(A) : A)[N,K] * W[K,F]          (xw, for edge scatter)
// G[N,F] = Y * nself[row] + bias[col]                 (agg init, self-loop)
template <int NT, bool RELU>
__global__ __launch_bounds__(128) void k_gemm_bf16(const float* __restrict__ A,
                                                   const v16bf* __restrict__ Wp,
                                                   const float* __restrict__ bias,
                                                   const float* __restrict__ nself,
                                                   float* __restrict__ Y,
                                                   float* __restrict__ G,
                                                   int mtiles, int K) {
  const int lane  = threadIdx.x & 31;
  const int mtile = blockIdx.x * 4 + (threadIdx.x >> 5);
  if (mtile >= mtiles) return;            // wave-uniform: EXEC stays all-ones
  const int Fout = NT * 16;
  const int m0   = mtile * 16;
  const int row  = m0 + (lane & 15);
  const int kb   = (lane >> 4) << 3;      // 0 or 8 (A/B K-split)
  const int hi   = (lane >> 4) << 3;      // row offset for C/D fragment

  v8f acc[NT] = {};

  for (int kk = 0; kk < K; kk += 32) {
    // A fragment: two aligned 32B runs (K kb..kb+7 and kb+16..kb+23)
    const float4* ap = (const float4*)(A + (size_t)row * K + kk + kb);
    float4 q0 = ap[0];
    float4 q1 = ap[1];
    float4 q2 = ap[4];
    float4 q3 = ap[5];
    float af[16] = {q0.x, q0.y, q0.z, q0.w, q1.x, q1.y, q1.z, q1.w,
                    q2.x, q2.y, q2.z, q2.w, q3.x, q3.y, q3.z, q3.w};
    v16bf a;
#pragma unroll
    for (int i = 0; i < 16; i++) {
      float f = af[i];
      if (RELU) f = fmaxf(f, 0.0f);
      a[i] = (__bf16)f;
    }
    // B fragments: contiguous 32B pre-packed loads
    const v16bf* bp = Wp + ((size_t)(kk >> 5) * NT) * 32 + lane;
#pragma unroll
    for (int t = 0; t < NT; t++) {
      v16bf b = bp[(size_t)t * 32];
      acc[t] = __builtin_amdgcn_wmma_f32_16x16x32_bf16(
          /*neg_a=*/false, a, /*neg_b=*/false, b,
          /*c_mod=*/(short)0, acc[t], /*reuse_a=*/false, /*reuse_b=*/false);
    }
  }

  // epilogue: write xw and agg-init = xw*nself + bias
  float nsv[8];
#pragma unroll
  for (int v = 0; v < 8; v++) nsv[v] = nself[m0 + v + hi];

#pragma unroll
  for (int t = 0; t < NT; t++) {
    const int   n  = t * 16 + (lane & 15);
    const float bn = bias[n];
#pragma unroll
    for (int v = 0; v < 8; v++) {
      const size_t idx = (size_t)(m0 + v + hi) * Fout + n;
      const float  y   = acc[t][v];
      Y[idx] = y;
      G[idx] = y * nsv[v] + bn;
    }
  }
}

// ---------------- edge scatter: agg[dst] += xw[src] * dinv[src]*dinv[dst] ----
__global__ __launch_bounds__(256) void k_scatter(const int* __restrict__ src,
                                                 const int* __restrict__ dst,
                                                 const float* __restrict__ dinv,
                                                 const float* __restrict__ xw,
                                                 float* __restrict__ agg,
                                                 int total, int cshift, int F) {
  int gid = blockIdx.x * blockDim.x + threadIdx.x;
  if (gid >= total) return;
  int e = gid >> cshift;
  int c = gid & ((1 << cshift) - 1);
  // edge indices / norms: streamed, no cross-pass reuse -> non-temporal
  int s = __builtin_nontemporal_load(src + e);
  int d = __builtin_nontemporal_load(dst + e);
  float w = __builtin_nontemporal_load(dinv + s) *
            __builtin_nontemporal_load(dinv + d);
  // xw gather: L2-resident reuse target -> default (RT) hint
  float4 x = ((const float4*)(xw + (size_t)s * F))[c];
  float* out = agg + (size_t)d * F + (c << 2);
  atomicAdd(out + 0, x.x * w);
  atomicAdd(out + 1, x.y * w);
  atomicAdd(out + 2, x.z * w);
  atomicAdd(out + 3, x.w * w);
}

// ---------------- GEMM dispatch helper ----------------
static inline void launch_gemm(int NT, int relu, const float* h, const __bf16* wpack,
                               const float* bias, const float* nself,
                               float* xw, float* agg, int mtiles, int K,
                               int gblocks, hipStream_t stream) {
  const v16bf* wp = (const v16bf*)wpack;
  if (relu) {
    switch (NT) {
      case 4:
        k_gemm_bf16<4, true><<<gblocks, 128, 0, stream>>>(h, wp, bias, nself, xw, agg, mtiles, K);
        break;
      case 8:
        k_gemm_bf16<8, true><<<gblocks, 128, 0, stream>>>(h, wp, bias, nself, xw, agg, mtiles, K);
        break;
      default:
        k_gemm_bf16<16, true><<<gblocks, 128, 0, stream>>>(h, wp, bias, nself, xw, agg, mtiles, K);
        break;
    }
  } else {
    switch (NT) {
      case 4:
        k_gemm_bf16<4, false><<<gblocks, 128, 0, stream>>>(h, wp, bias, nself, xw, agg, mtiles, K);
        break;
      case 8:
        k_gemm_bf16<8, false><<<gblocks, 128, 0, stream>>>(h, wp, bias, nself, xw, agg, mtiles, K);
        break;
      default:
        k_gemm_bf16<16, false><<<gblocks, 128, 0, stream>>>(h, wp, bias, nself, xw, agg, mtiles, K);
        break;
    }
  }
}

// ---------------- driver ----------------
extern "C" void kernel_launch(void* const* d_in, const int* in_sizes, int n_in,
                              void* d_out, int out_size, void* d_ws, size_t ws_size,
                              hipStream_t stream) {
  (void)n_in; (void)out_size; (void)ws_size;

  const float* x    = (const float*)d_in[0];
  const int*   ei   = (const int*)d_in[1];
  const int    E    = in_sizes[1] / 2;
  const int    N    = in_sizes[0] / 128;  // 50000
  const int*   src  = ei;
  const int*   dstp = ei + E;

  const float* Wm[6] = {(const float*)d_in[2], (const float*)d_in[4],
                        (const float*)d_in[6], (const float*)d_in[8],
                        (const float*)d_in[10], (const float*)d_in[12]};
  const float* Bm[6] = {(const float*)d_in[3], (const float*)d_in[5],
                        (const float*)d_in[7], (const float*)d_in[9],
                        (const float*)d_in[11], (const float*)d_in[13]};
  const int Kd[6] = {128, 256, 128, 64, 128, 256};
  const int Fd[6] = {256, 128, 64, 128, 256, 64};

  // workspace carve (256B aligned)
  uintptr_t wptr = (uintptr_t)d_ws;
  auto carve = [&](size_t bytes) -> void* {
    void* p = (void*)wptr;
    wptr += (bytes + 255) & ~(size_t)255;
    return p;
  };
  float*  deg   = (float*)carve((size_t)N * 4);
  float*  dinv  = (float*)carve((size_t)N * 4);
  float*  nself = (float*)carve((size_t)N * 4);
  __bf16* wpack = (__bf16*)carve((size_t)256 * 256 * 2);   // max K*F bf16
  float*  buf0  = (float*)carve((size_t)N * 256 * 4);      // xw scratch
  float*  buf1  = (float*)carve((size_t)N * 256 * 4);      // agg ping
  float*  buf2  = (float*)carve((size_t)N * 256 * 4);      // agg pong

  k_deg_init<<<(N + 255) / 256, 256, 0, stream>>>(deg, N);
  k_deg_count<<<(E + 255) / 256, 256, 0, stream>>>(dstp, deg, E);
  k_deg_fin<<<(N + 255) / 256, 256, 0, stream>>>(deg, dinv, nself, N);

  const int mtiles  = N / 16;               // 3125 (exact)
  const int gblocks = (mtiles + 3) / 4;     // 4 waves / block

  const float* h = x;
  float* aggs[6] = {buf1, buf2, buf1, buf2, buf1, (float*)d_out};

  for (int L = 0; L < 6; L++) {
    const int K = Kd[L], F = Fd[L];
    float* xw  = buf0;
    float* agg = aggs[L];

    // pack weights to bf16 fragment order (tiny: <= 4096 threads)
    const int wtotal = (K / 32) * (F / 16) * 32;
    k_pack_w<<<(wtotal + 255) / 256, 256, 0, stream>>>(Wm[L], wpack, K, F, wtotal);

    launch_gemm(F / 16, (L > 0) ? 1 : 0, h, wpack, Bm[L], nself, xw, agg,
                mtiles, K, gblocks, stream);

    const int cshift = (F == 256) ? 6 : (F == 128) ? 5 : 4;  // log2(F/4)
    const long long totalE = (long long)E << cshift;         // <= 102.4M, fits int
    k_scatter<<<(int)((totalE + 255) / 256), 256, 0, stream>>>(
        src, dstp, dinv, xw, agg, (int)totalE, cshift, F);

    h = agg;
  }
}
